// MotionDecoder_5866925326475
// MI455X (gfx1250) — matrix operand order repaired and checked by reading, not compiled, not run on hardware
//
#include <hip/hip_runtime.h>
#include <hip/hip_bf16.h>

// ---------------------------------------------------------------------------
// MI455X (gfx1250, wave32) implementation.
// Dominant GEMMs run on v_wmma_f32_16x16x32_f16; weights are pre-transposed
// to f16 (NxK) so A and B WMMA fragments load with contiguous b128s.
// Each wave computes a 32x64 C tile (2 A frags x 4 B frags -> 8 WMMAs/k-step,
// ~21 FLOP/byte of cache traffic) to stay matrix-pipe bound, not L2-bound.
// ---------------------------------------------------------------------------

typedef __attribute__((ext_vector_type(16))) _Float16 v16h;
typedef __attribute__((ext_vector_type(8)))  float    v8f;

union Frag16 { v16h v; uint4 q[2]; };

#define Bb   64
#define Tt   17
#define Ss   1024
#define HINc 256
#define Dd   768
#define NHh  12
#define HDd  64
#define NL   3
#define MT   (Bb*Tt)      // 1088 rows
#define STABc 0.001f
#define EPSc  0.001f
#define NEGc  (-1000000000.0f)

// ------------------------- generic WMMA GEMM -------------------------------
// C(M,N) = epi( A(M,K)f16 @ BT(N,K)^T f16 + bias )
// epi: 0 none, 1 relu, 2 relu+STAB, 3 split: col<splitN -> relu+STAB else none
// Block: 256 threads = 8 waves; wave w -> rows [by*256+32w, +32), cols [bx*64, +64)
__global__ void __launch_bounds__(256)
k_gemm_wmma(const _Float16* __restrict__ A, int lda,
            const _Float16* __restrict__ BT, int ldb,
            const float* __restrict__ bias,
            float* __restrict__ Cf, _Float16* __restrict__ Ch, int ldc,
            int M, int N, int K, int epi, int splitN)
{
    const int tid  = threadIdx.x;
    const int lane = tid & 31;
    const int wv   = tid >> 5;
    const int m0   = blockIdx.y * 256 + wv * 32;
    if (m0 >= M) return;
    const int n0   = blockIdx.x * 64;
    const int half = lane >> 4;   // 0 or 1
    const int l15  = lane & 15;

    v8f acc[8] = {};

    // A fragment: lane<16 holds row m+l15, K {0..15}+k0; lane>=16 K {16..31}
    const _Float16* aRow0 = A  + (size_t)(m0 +      l15) * lda + 16 * half;
    const _Float16* aRow1 = A  + (size_t)(m0 + 16 + l15) * lda + 16 * half;
    // B fragment: lane n = n0+16j+l15, contiguous K run of 16 at k0+16*half
    const _Float16* bRow0 = BT + (size_t)(n0 +  0 + l15) * ldb + 16 * half;
    const _Float16* bRow1 = BT + (size_t)(n0 + 16 + l15) * ldb + 16 * half;
    const _Float16* bRow2 = BT + (size_t)(n0 + 32 + l15) * ldb + 16 * half;
    const _Float16* bRow3 = BT + (size_t)(n0 + 48 + l15) * ldb + 16 * half;

    for (int k0 = 0; k0 < K; k0 += 32) {
        Frag16 a0, a1, b0, b1, b2, b3;
        a0.q[0] = *(const uint4*)(aRow0 + k0);  a0.q[1] = *(const uint4*)(aRow0 + k0 + 8);
        a1.q[0] = *(const uint4*)(aRow1 + k0);  a1.q[1] = *(const uint4*)(aRow1 + k0 + 8);
        b0.q[0] = *(const uint4*)(bRow0 + k0);  b0.q[1] = *(const uint4*)(bRow0 + k0 + 8);
        b1.q[0] = *(const uint4*)(bRow1 + k0);  b1.q[1] = *(const uint4*)(bRow1 + k0 + 8);
        b2.q[0] = *(const uint4*)(bRow2 + k0);  b2.q[1] = *(const uint4*)(bRow2 + k0 + 8);
        b3.q[0] = *(const uint4*)(bRow3 + k0);  b3.q[1] = *(const uint4*)(bRow3 + k0 + 8);

        acc[0] = __builtin_amdgcn_wmma_f32_16x16x32_f16(false, a0.v, false, b0.v, (short)0, acc[0], false, false);
        acc[1] = __builtin_amdgcn_wmma_f32_16x16x32_f16(false, a0.v, false, b1.v, (short)0, acc[1], false, false);
        acc[2] = __builtin_amdgcn_wmma_f32_16x16x32_f16(false, a0.v, false, b2.v, (short)0, acc[2], false, false);
        acc[3] = __builtin_amdgcn_wmma_f32_16x16x32_f16(false, a0.v, false, b3.v, (short)0, acc[3], false, false);
        acc[4] = __builtin_amdgcn_wmma_f32_16x16x32_f16(false, a1.v, false, b0.v, (short)0, acc[4], false, false);
        acc[5] = __builtin_amdgcn_wmma_f32_16x16x32_f16(false, a1.v, false, b1.v, (short)0, acc[5], false, false);
        acc[6] = __builtin_amdgcn_wmma_f32_16x16x32_f16(false, a1.v, false, b2.v, (short)0, acc[6], false, false);
        acc[7] = __builtin_amdgcn_wmma_f32_16x16x32_f16(false, a1.v, false, b3.v, (short)0, acc[7], false, false);
    }

    #pragma unroll
    for (int gI = 0; gI < 2; ++gI) {
        #pragma unroll
        for (int j = 0; j < 4; ++j) {
            int n = n0 + 16 * j + l15;
            float bv = bias ? bias[n] : 0.0f;
            bool stab = (epi == 2) || (epi == 3 && n < splitN);
            #pragma unroll
            for (int r = 0; r < 8; ++r) {
                float v = acc[gI * 4 + j][r] + bv;
                if (epi == 1)      v = fmaxf(v, 0.0f);
                else if (stab)     v = fmaxf(v, 0.0f) + STABc;
                size_t offc = (size_t)(m0 + 16 * gI + r + 8 * half) * ldc + n;
                if (Cf) Cf[offc] = v;
                if (Ch) Ch[offc] = (_Float16)v;
            }
        }
    }
}

// ------------------------- helpers -----------------------------------------
__global__ void k_cvt16(const float* __restrict__ in, _Float16* __restrict__ out, size_t n)
{
    size_t i = (size_t)blockIdx.x * blockDim.x + threadIdx.x;
    if (i < n) out[i] = (_Float16)in[i];
}

// in: (K,N) f32 row-major  ->  out: (N,K) f16 row-major
__global__ void k_transpose16(const float* __restrict__ in, _Float16* __restrict__ out, int K, int N)
{
    int k = blockIdx.x * 16 + threadIdx.x;
    int n = blockIdx.y * 16 + threadIdx.y;
    if (k < K && n < N) out[(size_t)n * K + k] = (_Float16)in[(size_t)k * N + n];
}

__global__ void k_addpos(float* __restrict__ x, _Float16* __restrict__ x16,
                         const float* __restrict__ pos)
{
    int i = blockIdx.x * blockDim.x + threadIdx.x;
    if (i >= MT * Dd) return;
    int bt = i / Dd, d = i % Dd, t = bt % Tt;
    float v = x[i] + pos[t * Dd + d];
    x[i] = v; x16[i] = (_Float16)v;
}

// ------------------------- self-attention (T=17) ----------------------------
__global__ void __launch_bounds__(64)
k_attn(const float* __restrict__ qh, const float* __restrict__ kh,
       const float* __restrict__ vh, const int* __restrict__ cmask,
       _Float16* __restrict__ o16)
{
    __shared__ float sq[Tt * 64], sk[Tt * 64], sv[Tt * 64], sp[Tt * Tt];
    int b = blockIdx.x / NHh, h = blockIdx.x % NHh;
    int tid = threadIdx.x;
    for (int i = tid; i < Tt * 64; i += 64) {
        int t = i >> 6, d = i & 63;
        size_t off = ((size_t)(b * Tt + t) * Dd) + h * 64 + d;
        sq[i] = qh[off]; sk[i] = kh[off]; sv[i] = vh[off];
    }
    __syncthreads();
    if (tid < Tt) {
        int tq = tid;
        float lg[Tt];
        float mx = -1e30f;
        #pragma unroll
        for (int tk = 0; tk < Tt; ++tk) {
            float s = 0.0f;
            #pragma unroll
            for (int d = 0; d < 64; ++d) s += sq[tq * 64 + d] * sk[tk * 64 + d];
            s *= 0.125f;                                   // 1/sqrt(64)
            bool ok = (tk <= tq) && (cmask[((size_t)b * Tt + tq) * Tt + tk] > 0);
            s = ok ? s : NEGc;
            lg[tk] = s;
            mx = fmaxf(mx, s);
        }
        float sum = 0.0f;
        #pragma unroll
        for (int tk = 0; tk < Tt; ++tk) { float e = __expf(lg[tk] - mx); lg[tk] = e; sum += e; }
        float inv = 1.0f / sum;
        #pragma unroll
        for (int tk = 0; tk < Tt; ++tk) sp[tq * Tt + tk] = lg[tk] * inv;
    }
    __syncthreads();
    for (int tq = 0; tq < Tt; ++tq) {
        float acc = 0.0f;
        #pragma unroll
        for (int tk = 0; tk < Tt; ++tk) acc += sp[tq * Tt + tk] * sv[tk * 64 + tid];
        o16[((size_t)(b * Tt + tq) * Dd) + h * 64 + tid] = (_Float16)acc;
    }
}

// ------------------------- residual layernorm -------------------------------
__global__ void __launch_bounds__(256)
k_ln_res(const float* __restrict__ a, const float* __restrict__ b,
         const float* __restrict__ g, const float* __restrict__ beta,
         float* outf, _Float16* outh)
{
    __shared__ float red[256];
    int row = blockIdx.x, tid = threadIdx.x;
    const float* pa = a + (size_t)row * Dd;
    const float* pb = b + (size_t)row * Dd;
    float x0 = pa[tid]       + pb[tid];
    float x1 = pa[tid + 256] + pb[tid + 256];
    float x2 = pa[tid + 512] + pb[tid + 512];
    red[tid] = x0 + x1 + x2;
    __syncthreads();
    for (int o = 128; o > 0; o >>= 1) { if (tid < o) red[tid] += red[tid + o]; __syncthreads(); }
    float mean = red[0] * (1.0f / 768.0f);
    __syncthreads();
    float d0 = x0 - mean, d1 = x1 - mean, d2 = x2 - mean;
    red[tid] = d0 * d0 + d1 * d1 + d2 * d2;
    __syncthreads();
    for (int o = 128; o > 0; o >>= 1) { if (tid < o) red[tid] += red[tid + o]; __syncthreads(); }
    float inv = rsqrtf(red[0] * (1.0f / 768.0f) + EPSc);
    float o0 = d0 * inv * g[tid]       + beta[tid];
    float o1 = d1 * inv * g[tid + 256] + beta[tid + 256];
    float o2 = d2 * inv * g[tid + 512] + beta[tid + 512];
    size_t base = (size_t)row * Dd;
    if (outf) { outf[base + tid] = o0; outf[base + tid + 256] = o1; outf[base + tid + 512] = o2; }
    if (outh) { outh[base + tid] = (_Float16)o0; outh[base + tid + 256] = (_Float16)o1; outh[base + tid + 512] = (_Float16)o2; }
}

// ------------- kv[e,f] = sum_s kp[s,e]*v[s,f]; kpsum[e] = sum_s kp[s,e] -----
// kpv: local chunk (bl, 1024, 1536) f16 ([0:768]=kp, [768:1536]=v)
__global__ void __launch_bounds__(256)
k_kv_reduce(const _Float16* __restrict__ kpv, float* __restrict__ kv,
            float* __restrict__ kpsum, int bbase)
{
    __shared__ _Float16 sh[8 * 128];
    int blk = blockIdx.x;
    int bl = blk / NHh, h = blk % NHh, b = bbase + bl;
    int tid = threadIdx.x;
    int e = tid >> 2;
    int f0 = (tid & 3) << 4;
    float acc[16] = {};
    float ksum = 0.0f;
    const _Float16* base = kpv + (size_t)bl * Ss * 1536;
    for (int sc = 0; sc < Ss; sc += 8) {
        __syncthreads();
        for (int i = tid; i < 8 * 128; i += 256) {
            int sr = i >> 7, c = i & 127;
            int col = (c < 64) ? (h * 64 + c) : (Dd + h * 64 + (c - 64));
            sh[i] = base[(size_t)(sc + sr) * 1536 + col];
        }
        __syncthreads();
        #pragma unroll
        for (int sr = 0; sr < 8; ++sr) {
            float a = (float)sh[sr * 128 + e];
            ksum += a;
            #pragma unroll
            for (int j = 0; j < 16; ++j) acc[j] += a * (float)sh[sr * 128 + 64 + f0 + j];
        }
    }
    size_t kvoff = (((size_t)(b * NHh + h)) * 64 + e) * 64 + f0;
    #pragma unroll
    for (int j = 0; j < 16; ++j) kv[kvoff + j] = acc[j];
    if (f0 == 0) kpsum[((size_t)(b * NHh + h)) * 64 + e] = ksum;
}

// ------------- r = (qp @ kv) / (qp . kpsum) --------------------------------
__global__ void __launch_bounds__(64)
k_linattn(const float* __restrict__ qp, const float* __restrict__ kv,
          const float* __restrict__ kpsum, _Float16* __restrict__ r16)
{
    __shared__ float sq[64], sks[64];
    int b = blockIdx.x / NHh, h = blockIdx.x % NHh;
    int f = threadIdx.x;
    sks[f] = kpsum[((size_t)(b * NHh + h)) * 64 + f];
    float kcol[64];
    #pragma unroll
    for (int e = 0; e < 64; ++e)
        kcol[e] = kv[(((size_t)(b * NHh + h)) * 64 + e) * 64 + f];
    __syncthreads();
    for (int t = 0; t < Tt; ++t) {
        __syncthreads();
        sq[f] = qp[((size_t)(b * Tt + t) * Dd) + h * 64 + f];
        __syncthreads();
        float num = 0.0f, den = 0.0f;
        #pragma unroll
        for (int e = 0; e < 64; ++e) { num += sq[e] * kcol[e]; den += sq[e] * sks[e]; }
        r16[((size_t)(b * Tt + t) * Dd) + h * 64 + f] = (_Float16)(num / den);
    }
}

// ------------- maxpool groups of 4 along D ---------------------------------
__global__ void k_maxpool(const float* __restrict__ x, _Float16* __restrict__ z)
{
    int i = blockIdx.x * blockDim.x + threadIdx.x;
    if (i >= Bb * Tt * (Dd / 4)) return;
    int b = i / (Tt * (Dd / 4));
    int rem = i % (Tt * (Dd / 4));
    int t = rem / (Dd / 4), j = rem % (Dd / 4);
    const float* p = x + ((size_t)(b * Tt + t) * Dd) + j * 4;
    float m = fmaxf(fmaxf(p[0], p[1]), fmaxf(p[2], p[3]));
    z[(size_t)b * (Tt * (Dd / 4)) + t * (Dd / 4) + j] = (_Float16)m;
}

// ------------------------- host side ---------------------------------------
static inline void gemm(hipStream_t s, const _Float16* A, int lda, const _Float16* BT,
                        const float* bias, float* Cf, _Float16* Ch, int ldc,
                        int M, int N, int K, int epi, int splitN = 0)
{
    dim3 g(N / 64, (M + 255) / 256), b(256);
    k_gemm_wmma<<<g, b, 0, s>>>(A, lda, BT, K, bias, Cf, Ch, ldc, M, N, K, epi, splitN);
}

static inline void launch_T(hipStream_t s, const float* in, _Float16* out, int K, int N)
{
    dim3 g((K + 15) / 16, (N + 15) / 16), b(16, 16);
    k_transpose16<<<g, b, 0, s>>>(in, out, K, N);
}

static inline void launch_cvt(hipStream_t s, const float* in, _Float16* out, size_t n)
{
    k_cvt16<<<(unsigned)((n + 255) / 256), 256, 0, s>>>(in, out, n);
}

extern "C" void kernel_launch(void* const* d_in, const int* in_sizes, int n_in,
                              void* d_out, int out_size, void* d_ws, size_t ws_size,
                              hipStream_t stream)
{
    const float* inputs = (const float*)d_in[0];
    const float* scene  = (const float*)d_in[1];
    const int*   cmask  = (const int*)d_in[2];
    const float* w1     = (const float*)d_in[3];
    const float* b1     = (const float*)d_in[4];
    const float* pos    = (const float*)d_in[5];
    const float* wqkv   = (const float*)d_in[6];
    const float* bqkv   = (const float*)d_in[7];
    const float* sa_wq  = (const float*)d_in[8];
    const float* sa_bq  = (const float*)d_in[9];
    const float* sa_wk  = (const float*)d_in[10];
    const float* sa_bk  = (const float*)d_in[11];
    const float* sa_wv  = (const float*)d_in[12];
    const float* sa_bv  = (const float*)d_in[13];
    const float* sa_wo  = (const float*)d_in[14];
    const float* sa_bo  = (const float*)d_in[15];
    const float* ca_wq  = (const float*)d_in[16];
    const float* ca_wk  = (const float*)d_in[17];
    const float* ca_wv  = (const float*)d_in[18];
    const float* ca_wo  = (const float*)d_in[19];
    const float* ln1_g  = (const float*)d_in[20];
    const float* ln1_b  = (const float*)d_in[21];
    const float* ln2_g  = (const float*)d_in[22];
    const float* ln2_b  = (const float*)d_in[23];
    const float* ln3_g  = (const float*)d_in[24];
    const float* ln3_b  = (const float*)d_in[25];
    const float* fc_w   = (const float*)d_in[26];
    const float* fc_b   = (const float*)d_in[27];
    const float* w2     = (const float*)d_in[28];
    const float* b2     = (const float*)d_in[29];
    float* out = (float*)d_out;

    // ---------------- workspace bump allocator ----------------
    char* wsp = (char*)d_ws;
    size_t off = 0;
    auto alloc = [&](size_t bytes) -> char* {
        char* p = wsp + off;
        off = (off + bytes + 255) & ~(size_t)255;
        return p;
    };

    _Float16* scene16 = (_Float16*)alloc((size_t)Bb * Ss * Dd * 2);
    _Float16* in16    = (_Float16*)alloc((size_t)MT * HINc * 2);
    _Float16* w1T     = (_Float16*)alloc((size_t)Dd * HINc * 2);
    _Float16* wqkvT   = (_Float16*)alloc((size_t)3 * Dd * Dd * 2);
    _Float16* w2T     = (_Float16*)alloc((size_t)1024 * (Tt * Dd / 4) * 2);
    _Float16* saqT    = (_Float16*)alloc((size_t)NL * Dd * Dd * 2);
    _Float16* sakT    = (_Float16*)alloc((size_t)NL * Dd * Dd * 2);
    _Float16* savT    = (_Float16*)alloc((size_t)NL * Dd * Dd * 2);
    _Float16* sawoT   = (_Float16*)alloc((size_t)NL * Dd * Dd * 2);
    _Float16* cawqT   = (_Float16*)alloc((size_t)NL * Dd * Dd * 2);
    _Float16* kvT     = (_Float16*)alloc((size_t)NL * 1536 * Dd * 2);
    _Float16* cawoT   = (_Float16*)alloc((size_t)NL * Dd * Dd * 2);
    _Float16* fcT     = (_Float16*)alloc((size_t)NL * Dd * Dd * 2);

    float*    xf      = (float*)   alloc((size_t)MT * Dd * 4);
    _Float16* x16     = (_Float16*)alloc((size_t)MT * Dd * 2);
    _Float16* qkv16   = (_Float16*)alloc((size_t)MT * 3 * Dd * 2);
    float*    qhf     = (float*)   alloc((size_t)MT * Dd * 4);
    float*    khf     = (float*)   alloc((size_t)MT * Dd * 4);
    float*    vhf     = (float*)   alloc((size_t)MT * Dd * 4);
    _Float16* o16     = (_Float16*)alloc((size_t)MT * Dd * 2);
    float*    a1f     = (float*)   alloc((size_t)MT * Dd * 4);
    _Float16* qn16    = (_Float16*)alloc((size_t)MT * Dd * 2);
    float*    qpf     = (float*)   alloc((size_t)MT * Dd * 4);
    const int CHB = 8;                 // batches per scene chunk
    _Float16* kpv16   = (_Float16*)alloc((size_t)CHB * Ss * 1536 * 2);
    float*    kvbuf   = (float*)   alloc((size_t)Bb * NHh * 64 * 64 * 4);
    float*    kpsum   = (float*)   alloc((size_t)Bb * NHh * 64 * 4);
    _Float16* r16     = (_Float16*)alloc((size_t)MT * Dd * 2);
    float*    a2f     = (float*)   alloc((size_t)MT * Dd * 4);
    float*    nrmf    = (float*)   alloc((size_t)MT * Dd * 4);
    _Float16* nrm16   = (_Float16*)alloc((size_t)MT * Dd * 2);
    float*    yf      = (float*)   alloc((size_t)MT * Dd * 4);
    _Float16* z16     = (_Float16*)alloc((size_t)Bb * (Tt * Dd / 4) * 2);
    (void)ws_size; (void)n_in; (void)in_sizes; (void)out_size;

    // ---------------- prepass: f16 converts + weight transposes -------------
    launch_cvt(stream, scene,  scene16, (size_t)Bb * Ss * Dd);
    launch_cvt(stream, inputs, in16,    (size_t)MT * HINc);
    launch_T(stream, w1,   w1T,   HINc, Dd);
    launch_T(stream, wqkv, wqkvT, Dd,   3 * Dd);
    launch_T(stream, w2,   w2T,   Tt * Dd / 4, 1024);
    for (int i = 0; i < NL; ++i) {
        size_t wo = (size_t)i * Dd * Dd;
        launch_T(stream, sa_wq + wo, saqT  + wo, Dd, Dd);
        launch_T(stream, sa_wk + wo, sakT  + wo, Dd, Dd);
        launch_T(stream, sa_wv + wo, savT  + wo, Dd, Dd);
        launch_T(stream, sa_wo + wo, sawoT + wo, Dd, Dd);
        launch_T(stream, ca_wq + wo, cawqT + wo, Dd, Dd);
        launch_T(stream, ca_wk + wo, kvT + (size_t)i * 1536 * Dd,                   Dd, Dd);
        launch_T(stream, ca_wv + wo, kvT + (size_t)i * 1536 * Dd + (size_t)Dd * Dd, Dd, Dd);
        launch_T(stream, ca_wo + wo, cawoT + wo, Dd, Dd);
        launch_T(stream, fc_w  + wo, fcT   + wo, Dd, Dd);
    }

    // ---------------- embed: x = relu(inputs@w1 + b1) + pos -----------------
    gemm(stream, in16, HINc, w1T, b1, xf, nullptr, Dd, MT, Dd, HINc, /*relu*/1);
    k_addpos<<<(MT * Dd + 255) / 256, 256, 0, stream>>>(xf, x16, pos);

    // ---------------- layers ------------------------------------------------
    for (int i = 0; i < NL; ++i) {
        size_t wo = (size_t)i * Dd * Dd;

        // qkv = relu(x @ wqkv + bqkv)   (f16 only; feeds head projections)
        gemm(stream, x16, Dd, wqkvT, bqkv, nullptr, qkv16, 3 * Dd, MT, 3 * Dd, Dd, 1);

        // per-head projections
        gemm(stream, qkv16,          3 * Dd, saqT + wo, sa_bq + (size_t)i * Dd, qhf, nullptr, Dd, MT, Dd, Dd, 0);
        gemm(stream, qkv16 + Dd,     3 * Dd, sakT + wo, sa_bk + (size_t)i * Dd, khf, nullptr, Dd, MT, Dd, Dd, 0);
        gemm(stream, qkv16 + 2 * Dd, 3 * Dd, savT + wo, sa_bv + (size_t)i * Dd, vhf, nullptr, Dd, MT, Dd, Dd, 0);

        // masked softmax attention (T=17)
        k_attn<<<Bb * NHh, 64, 0, stream>>>(qhf, khf, vhf, cmask, o16);

        // a1 = o @ sa_wo + sa_bo ; qn = LN(a1 + x)
        gemm(stream, o16, Dd, sawoT + wo, sa_bo + (size_t)i * Dd, a1f, nullptr, Dd, MT, Dd, Dd, 0);
        k_ln_res<<<MT, 256, 0, stream>>>(a1f, xf, ln1_g + (size_t)i * Dd, ln1_b + (size_t)i * Dd,
                                         nullptr, qn16);

        // qp = relu(qn @ ca_wq) + STAB
        gemm(stream, qn16, Dd, cawqT + wo, nullptr, qpf, nullptr, Dd, MT, Dd, Dd, /*relu+stab*/2);

        // scene projections (the big GEMM) + kv/kpsum reduction, chunked
        for (int c = 0; c < Bb / CHB; ++c) {
            const _Float16* Ab = scene16 + (size_t)c * CHB * Ss * Dd;
            gemm(stream, Ab, Dd, kvT + (size_t)i * 1536 * Dd, nullptr,
                 nullptr, kpv16, 1536, CHB * Ss, 1536, Dd, /*split*/3, /*splitN*/Dd);
            k_kv_reduce<<<CHB * NHh, 256, 0, stream>>>(kpv16, kvbuf, kpsum, c * CHB);
        }

        // r = (qp @ kv) / (qp . kpsum)
        k_linattn<<<Bb * NHh, 64, 0, stream>>>(qpf, kvbuf, kpsum, r16);

        // a2 = r @ ca_wo ; nrm = LN(a2 + x)
        gemm(stream, r16, Dd, cawoT + wo, nullptr, a2f, nullptr, Dd, MT, Dd, Dd, 0);
        k_ln_res<<<MT, 256, 0, stream>>>(a2f, xf, ln2_g + (size_t)i * Dd, ln2_b + (size_t)i * Dd,
                                         nrmf, nrm16);

        // y = relu(nrm @ fc_w + fc_b) ; x = LN(nrm + y)
        gemm(stream, nrm16, Dd, fcT + wo, fc_b + (size_t)i * Dd, yf, nullptr, Dd, MT, Dd, Dd, 1);
        k_ln_res<<<MT, 256, 0, stream>>>(nrmf, yf, ln3_g + (size_t)i * Dd, ln3_b + (size_t)i * Dd,
                                         xf, x16);
    }

    // ---------------- head: maxpool(4) -> relu(z @ w2 + b2) -----------------
    k_maxpool<<<(Bb * Tt * (Dd / 4) + 255) / 256, 256, 0, stream>>>(xf, z16);
    gemm(stream, z16, Tt * Dd / 4, w2T, b2, out, nullptr, 1024, Bb, 1024, Tt * Dd / 4, 1);
}
